// CumsumAttention_45629732553265
// MI455X (gfx1250) — matrix operand order repaired and checked by reading, not compiled, not run on hardware
//
#include <hip/hip_runtime.h>
#include <math.h>

typedef __bf16 bf16_t;
typedef __attribute__((ext_vector_type(16))) __bf16 v16bf;
typedef __attribute__((ext_vector_type(8)))  float  v8f;

static __device__ inline v8f wmma_bf16(v16bf a, v16bf b, v8f c) {
    // D = A(16x32 bf16) x B(32x16 bf16) + C(16x16 f32)
    return __builtin_amdgcn_wmma_f32_16x16x32_bf16(
        /*neg_a=*/false, a, /*neg_b=*/false, b,
        /*c_mod=*/(short)0, c, /*reuse_a=*/false, /*reuse_b=*/false);
}

static __device__ inline v8f zero8() {
    v8f v;
    #pragma unroll
    for (int i = 0; i < 8; ++i) v[i] = 0.0f;
    return v;
}

// A-fragment K index for element i, lane-half h (ISA 7.12.2, 16-bit A 16x32)
static __device__ inline int a_kidx(int i, int h) {
    return h * 8 + (i < 8 ? i : i + 8);
}

// ---------------------------------------------------------------------------
// fp32 -> bf16 elementwise convert
// ---------------------------------------------------------------------------
__global__ void cvt_f32_bf16_kernel(const float* __restrict__ in,
                                    bf16_t* __restrict__ out, long n) {
    long i = (long)blockIdx.x * blockDim.x + threadIdx.x;
    if (i < n) out[i] = (bf16_t)in[i];
}

// ---------------------------------------------------------------------------
// fp32 [N][K] -> bf16 transposed [K][N]   (one-time weight prep; W reused 256x)
// ---------------------------------------------------------------------------
__global__ __launch_bounds__(256) void cvtT_f32_bf16_kernel(
        const float* __restrict__ in, bf16_t* __restrict__ out, int N, int K) {
    __shared__ float tile[32][33];
    int n0 = blockIdx.x * 32, k0 = blockIdx.y * 32;
    int tx = threadIdx.x, ty = threadIdx.y;
    #pragma unroll
    for (int j = 0; j < 4; ++j)
        tile[ty + 8 * j][tx] = in[(size_t)(n0 + ty + 8 * j) * K + k0 + tx];
    __syncthreads();
    #pragma unroll
    for (int j = 0; j < 4; ++j)
        out[(size_t)(k0 + ty + 8 * j) * N + n0 + tx] = (bf16_t)tile[tx][ty + 8 * j];
}

// ---------------------------------------------------------------------------
// bf16 [B][T][C] -> bf16 [B][C][T]   (q transpose for the wei GEMM B operand)
// ---------------------------------------------------------------------------
__global__ __launch_bounds__(256) void transpose_bf16_kernel(
        const bf16_t* __restrict__ in, bf16_t* __restrict__ out, int T, int C) {
    __shared__ bf16_t tile[32][33];
    int b = blockIdx.x, t0 = blockIdx.y * 32, c0 = blockIdx.z * 32;
    int tx = threadIdx.x, ty = threadIdx.y;
    const bf16_t* ib = in + (size_t)b * T * C;
    bf16_t* ob = out + (size_t)b * T * C;
    #pragma unroll
    for (int j = 0; j < 4; ++j)
        tile[ty + 8 * j][tx] = ib[(size_t)(t0 + ty + 8 * j) * C + c0 + tx];
    __syncthreads();
    #pragma unroll
    for (int j = 0; j < 4; ++j)
        ob[(size_t)(c0 + ty + 8 * j) * T + t0 + tx] = tile[tx][ty + 8 * j];
}

// ---------------------------------------------------------------------------
// Reversed cumulative angle: angle[b,t,c] = sum_{s>=t} emb[idx[b,s], c]
// ---------------------------------------------------------------------------
__global__ void angle_kernel(const int* __restrict__ idx,
                             const float* __restrict__ emb,
                             float* __restrict__ cosA,
                             float* __restrict__ sinA,
                             int T, int Ch) {
    int b = blockIdx.x;
    int c = blockIdx.y * blockDim.x + threadIdx.x;
    float acc = 0.0f;
    for (int t = T - 1; t >= 0; --t) {
        int tok = idx[b * T + t];
        acc += emb[(size_t)tok * Ch + c];
        size_t o = ((size_t)b * T + t) * Ch + c;
        cosA[o] = cosf(acc);
        sinA[o] = sinf(acc);
    }
}

// ---------------------------------------------------------------------------
// Channel-pair rotation (fwd: [[c,-s],[s,c]], inv: transpose), f32 -> bf16
// ---------------------------------------------------------------------------
__global__ void rot_kernel(const float* __restrict__ y,
                           const float* __restrict__ cosA,
                           const float* __restrict__ sinA,
                           bf16_t* __restrict__ out, long npairs, int inv) {
    long i = (long)blockIdx.x * blockDim.x + threadIdx.x;
    if (i >= npairs) return;
    float z0 = y[2 * i], z1 = y[2 * i + 1];
    float c = cosA[i], s = sinA[i];
    float o0, o1;
    if (inv) { o0 = c * z0 + s * z1; o1 = -s * z0 + c * z1; }
    else     { o0 = c * z0 - s * z1; o1 =  s * z0 + c * z1; }
    out[2 * i]     = (bf16_t)o0;
    out[2 * i + 1] = (bf16_t)o1;
}

// ---------------------------------------------------------------------------
// Tiled WMMA GEMM:  Y[m,n] = sum_k A[m,k] * Wt[k,n] + bias[n]
// A: [M,K] bf16, Wt: [K,N] bf16 (pre-transposed), Y: [M,N] f32.
// Block tile 128x128, 256 threads = 8 waves, each wave 32x64 = 2x4 WMMA tiles.
// Tiles staged via CDNA5 global_load_async_to_lds_b128, DOUBLE-BUFFERED:
// tile i+1 streams while tile i is computed; async loads complete in order,
// so s_wait_asynccnt 4 (= the 4 just-issued next-tile ops) gates tile i.
// The LDS base pointer is passed to the asm (ptrtoint = capture) so the
// compiler cannot DCE the ds_load fragment reads.
// ---------------------------------------------------------------------------
#define GT_M 128
#define GT_N 128
#define GT_K 32
#define GT_BUF (GT_M * GT_K + GT_K * GT_N)   // elements per stage buffer

__global__ __launch_bounds__(256) void gemm_bias_kernel(
        const bf16_t* __restrict__ A, const bf16_t* __restrict__ Wt,
        const float* __restrict__ bias, float* __restrict__ Y,
        int M, int N, int K) {
    __shared__ __align__(16) bf16_t smem[2 * GT_BUF];   // arena at LDS offset 0

    const int tid  = threadIdx.x;
    const int wave = tid >> 5;
    const int lane = tid & 31;
    const int r    = lane & 15;
    const int h    = lane >> 4;

    const int mBase = blockIdx.x * GT_M;
    const int nBase = blockIdx.y * GT_N;
    const int wr = (wave & 3) * 32;    // wave row offset
    const int wc = (wave >> 2) * 64;   // wave col offset

    // per-thread async-copy assignments (16 bf16 = 32 bytes each, 2x b128)
    const int arow = tid >> 1, acol = (tid & 1) * 16;   // A: 128 rows x 2 halves
    const int brow = tid >> 3, bcol = (tid & 7) * 16;   // B: 32 rows x 8 segs
    const unsigned offA = (unsigned)(arow * GT_K + acol) * 2u;
    const unsigned offB = (unsigned)(GT_M * GT_K + brow * GT_N + bcol) * 2u;
    // escape the LDS arena so the asm is known to write it
    const unsigned long long esc = (unsigned long long)(size_t)&smem[0];

    auto issue_tile = [&](int k0, int buf) {
        unsigned la = (unsigned)buf * (GT_BUF * 2u) + offA;
        unsigned lb = (unsigned)buf * (GT_BUF * 2u) + offB;
        unsigned long long ga =
            (unsigned long long)(A + (size_t)(mBase + arow) * K + k0 + acol);
        unsigned long long gb =
            (unsigned long long)(Wt + (size_t)(k0 + brow) * N + nBase + bcol);
        // offset: is added to both the LDS and the global address
        asm volatile("global_load_async_to_lds_b128 %0, %1, off"
                     :: "v"(la), "v"(ga), "v"(esc) : "memory");
        asm volatile("global_load_async_to_lds_b128 %0, %1, off offset:16"
                     :: "v"(la), "v"(ga), "v"(esc) : "memory");
        asm volatile("global_load_async_to_lds_b128 %0, %1, off"
                     :: "v"(lb), "v"(gb), "v"(esc) : "memory");
        asm volatile("global_load_async_to_lds_b128 %0, %1, off offset:16"
                     :: "v"(lb), "v"(gb), "v"(esc) : "memory");
    };

    v8f acc[2][4];
    #pragma unroll
    for (int ti = 0; ti < 2; ++ti)
        #pragma unroll
        for (int tj = 0; tj < 4; ++tj) acc[ti][tj] = zero8();

    const int nIter = K / GT_K;
    issue_tile(0, 0);

    for (int it = 0; it < nIter; ++it) {
        const int buf = it & 1;
        if (it + 1 < nIter) {
            // other buffer's readers finished before the barrier ending it-1
            issue_tile((it + 1) * GT_K, buf ^ 1);
            asm volatile("s_wait_asynccnt 0x4" ::: "memory");  // tile `it` landed
        } else {
            asm volatile("s_wait_asynccnt 0x0" ::: "memory");
        }
        __syncthreads();

        const bf16_t* As = smem + buf * GT_BUF;              // [GT_M][GT_K]
        const bf16_t* Bs = As + GT_M * GT_K;                 // [GT_K][GT_N]

        v16bf afr[2], bfr[4];
        #pragma unroll
        for (int ti = 0; ti < 2; ++ti) {
            int row = wr + ti * 16 + r;
            #pragma unroll
            for (int i = 0; i < 16; ++i) afr[ti][i] = As[row * GT_K + a_kidx(i, h)];
        }
        #pragma unroll
        for (int tj = 0; tj < 4; ++tj) {
            int col = wc + tj * 16;
            #pragma unroll
            for (int i = 0; i < 16; ++i) bfr[tj][i] = Bs[a_kidx(i, h) * GT_N + col + r];
        }
        #pragma unroll
        for (int ti = 0; ti < 2; ++ti)
            #pragma unroll
            for (int tj = 0; tj < 4; ++tj)
                acc[ti][tj] = wmma_bf16(afr[ti], bfr[tj], acc[ti][tj]);
        __syncthreads();   // all reads of `buf` done before it is refilled
    }

    // epilogue: D tile lane holds (M = p + 8h, N = r)
    #pragma unroll
    for (int ti = 0; ti < 2; ++ti)
        #pragma unroll
        for (int tj = 0; tj < 4; ++tj) {
            int n = nBase + wc + tj * 16 + r;
            float bv = bias[n];
            #pragma unroll
            for (int p = 0; p < 8; ++p) {
                int m = mBase + wr + ti * 16 + p + 8 * h;
                Y[(size_t)m * N + n] = acc[ti][tj][p] + bv;
            }
        }
}

// ---------------------------------------------------------------------------
// Fused causal softplus attention (no normalization -> single streaming pass):
//   out[t,:] = sum_{s<=t} softplus( (k[t]·q[s]) * scale ) * v[s,:]
// Block = (batch b, 16-row t-tile). 8 waves; wave w owns channel slice
// [w*128, w*128+128) for both the k·q^T partial reduction and out columns.
// q is pre-transposed to [B][C][T] so the B-fragment is a contiguous load.
// ---------------------------------------------------------------------------
__global__ __launch_bounds__(256) void attn_kernel(
        const bf16_t* __restrict__ kh, const bf16_t* __restrict__ qT,
        const bf16_t* __restrict__ vh, float* __restrict__ out,
        int T, int C, float scale) {
    const int b     = blockIdx.x;
    const int tBase = blockIdx.y * 16;
    const int tid   = threadIdx.x;
    const int wave  = tid >> 5;
    const int lane  = tid & 31;
    const int r     = lane & 15;
    const int h     = lane >> 4;
    const int cw    = wave * 128;

    __shared__ float  weiF[16 * 32];   // f32 reduction buffer  [t_local][s_local]
    __shared__ bf16_t weiH[16 * 32];   // bf16 A-operand for stage 2

    const bf16_t* kb  = kh + (size_t)b * T * C;
    const bf16_t* qTb = qT + (size_t)b * T * C;   // [C][T]
    const bf16_t* vb  = vh + (size_t)b * T * C;

    v8f outAcc[8];
    #pragma unroll
    for (int i = 0; i < 8; ++i) outAcc[i] = zero8();

    const int sEnd = tBase + 16;   // causal: s <= tBase+15
    for (int sBase = 0; sBase < sEnd; sBase += 32) {
        weiF[tid] = 0.0f;
        weiF[tid + 256] = 0.0f;
        __syncthreads();

        // ---- stage 1: partial wei over this wave's 128-channel slice
        v8f w0 = zero8(), w1 = zero8();
        for (int kc = 0; kc < 4; ++kc) {
            int c0 = cw + kc * 32;
            v16bf a;
            const bf16_t* krow = kb + (size_t)(tBase + r) * C + c0;
            #pragma unroll
            for (int i = 0; i < 16; ++i) a[i] = krow[a_kidx(i, h)];
            // B frag: B[k=channel=a_kidx(i,h)][n=s_local=r] = qT[c0+k][sBase+n]
            const bf16_t* qcol = qTb + (size_t)c0 * T + sBase + r;
            v16bf b0, b1;
            #pragma unroll
            for (int i = 0; i < 16; ++i) {
                b0[i] = qcol[(size_t)a_kidx(i, h) * T];
                b1[i] = qcol[(size_t)a_kidx(i, h) * T + 16];
            }
            w0 = wmma_bf16(a, b0, w0);
            w1 = wmma_bf16(a, b1, w1);
        }
        // cross-wave reduction: D tile lane holds (t = p + 8h, s = r)
        #pragma unroll
        for (int p = 0; p < 8; ++p) {
            atomicAdd(&weiF[(p + 8 * h) * 32 + r],      w0[p]);
            atomicAdd(&weiF[(p + 8 * h) * 32 + 16 + r], w1[p]);
        }
        __syncthreads();

        // ---- scale + causal mask + softplus, write bf16 operand
        #pragma unroll
        for (int e = tid; e < 512; e += 256) {
            int tl = e >> 5, sl = e & 31;
            int tg = tBase + tl, sg = sBase + sl;
            float x = weiF[e] * scale;
            float y = 0.0f;
            if (sg <= tg)
                y = (x > 20.0f) ? x : log1pf(expf(x));   // softplus
            weiH[e] = (bf16_t)y;
        }
        __syncthreads();

        // ---- stage 2: out += wei(16x32) x v(32 x 128-slice)
        v16bf a2;
        #pragma unroll
        for (int i = 0; i < 16; ++i) a2[i] = weiH[r * 32 + a_kidx(i, h)];
        const bf16_t* vrow = vb + (size_t)sBase * C + cw + r;
        #pragma unroll
        for (int ct = 0; ct < 8; ++ct) {
            v16bf bf;
            #pragma unroll
            for (int i = 0; i < 16; ++i) bf[i] = vrow[(size_t)a_kidx(i, h) * C + ct * 16];
            outAcc[ct] = wmma_bf16(a2, bf, outAcc[ct]);
        }
        __syncthreads();
    }

    #pragma unroll
    for (int ct = 0; ct < 8; ++ct)
        #pragma unroll
        for (int p = 0; p < 8; ++p) {
            int tt = tBase + p + 8 * h;
            int cc = cw + ct * 16 + r;
            out[((size_t)b * T + tt) * C + cc] = outAcc[ct][p];
        }
}

// ---------------------------------------------------------------------------
// Host-side orchestration
// ---------------------------------------------------------------------------
extern "C" void kernel_launch(void* const* d_in, const int* in_sizes, int n_in,
                              void* d_out, int out_size, void* d_ws, size_t ws_size,
                              hipStream_t stream) {
    (void)in_sizes; (void)n_in; (void)out_size; (void)ws_size;
    const int Bn = 16, T = 1024, C = 1024, Ch = C / 2;
    const long M = (long)Bn * T;            // 16384 rows

    const float* x    = (const float*)d_in[0];
    const int*   idx  = (const int*)  d_in[1];
    const float* Wk   = (const float*)d_in[2];
    const float* bk   = (const float*)d_in[3];
    const float* Wq   = (const float*)d_in[4];
    const float* bq   = (const float*)d_in[5];
    const float* Wv   = (const float*)d_in[6];
    const float* bv   = (const float*)d_in[7];
    const float* Wp   = (const float*)d_in[8];
    const float* bp   = (const float*)d_in[9];
    const float* ang  = (const float*)d_in[10];

    // ---- workspace carve-up
    char* p = (char*)d_ws;
    bf16_t* xh   = (bf16_t*)p; p += (size_t)M * C * sizeof(bf16_t);   // 32 MB
    bf16_t* Wkt  = (bf16_t*)p; p += (size_t)C * C * sizeof(bf16_t);   // [K][N]
    bf16_t* Wqt  = (bf16_t*)p; p += (size_t)C * C * sizeof(bf16_t);
    bf16_t* Wvt  = (bf16_t*)p; p += (size_t)C * C * sizeof(bf16_t);
    bf16_t* Wpt  = (bf16_t*)p; p += (size_t)C * C * sizeof(bf16_t);
    float*  cosA = (float*)p; p += (size_t)M * Ch * sizeof(float);    // 32 MB
    float*  sinA = (float*)p; p += (size_t)M * Ch * sizeof(float);    // 32 MB
    float*  tmp  = (float*)p; p += (size_t)M * C * sizeof(float);     // 64 MB
    bf16_t* kh   = (bf16_t*)p; p += (size_t)M * C * sizeof(bf16_t);
    bf16_t* qh   = (bf16_t*)p; p += (size_t)M * C * sizeof(bf16_t);
    bf16_t* vh   = (bf16_t*)p; p += (size_t)M * C * sizeof(bf16_t);
    bf16_t* ah   = (bf16_t*)p; p += (size_t)M * C * sizeof(bf16_t);
    bf16_t* qT   = (bf16_t*)p; p += (size_t)M * C * sizeof(bf16_t);   // [B][C][T]

    const float scale = 1.0f / sqrtf((float)C);

    // 1) converts: x -> bf16; weights -> bf16 transposed [K][N]
    {
        long n = M * C;
        cvt_f32_bf16_kernel<<<dim3((n + 255) / 256), dim3(256), 0, stream>>>(x, xh, n);
        dim3 tg(C / 32, C / 32), tb(32, 8);
        cvtT_f32_bf16_kernel<<<tg, tb, 0, stream>>>(Wk, Wkt, C, C);
        cvtT_f32_bf16_kernel<<<tg, tb, 0, stream>>>(Wq, Wqt, C, C);
        cvtT_f32_bf16_kernel<<<tg, tb, 0, stream>>>(Wv, Wvt, C, C);
        cvtT_f32_bf16_kernel<<<tg, tb, 0, stream>>>(Wp, Wpt, C, C);
    }

    // 2) reversed-cumsum angles -> cos/sin
    angle_kernel<<<dim3(Bn, Ch / 128), dim3(128), 0, stream>>>(idx, ang, cosA, sinA, T, Ch);

    // 3) q/k/v projections (GEMM -> f32 tmp, rotate -> bf16)
    dim3 ggrid(M / GT_M, C / GT_N);
    long npairs = M * Ch;
    dim3 rgrid((npairs + 255) / 256);

    gemm_bias_kernel<<<ggrid, dim3(256), 0, stream>>>(xh, Wkt, bk, tmp, (int)M, C, C);
    rot_kernel<<<rgrid, dim3(256), 0, stream>>>(tmp, cosA, sinA, kh, npairs, 0);

    gemm_bias_kernel<<<ggrid, dim3(256), 0, stream>>>(xh, Wqt, bq, tmp, (int)M, C, C);
    rot_kernel<<<rgrid, dim3(256), 0, stream>>>(tmp, cosA, sinA, qh, npairs, 0);

    gemm_bias_kernel<<<ggrid, dim3(256), 0, stream>>>(xh, Wvt, bv, tmp, (int)M, C, C);
    rot_kernel<<<rgrid, dim3(256), 0, stream>>>(tmp, cosA, sinA, vh, npairs, 0);

    // 3b) q -> q^T  [B][C][T]
    transpose_bf16_kernel<<<dim3(Bn, T / 32, C / 32), dim3(32, 8), 0, stream>>>(qh, qT, T, C);

    // 4) fused causal-softplus attention -> f32 tmp
    attn_kernel<<<dim3(Bn, T / 16), dim3(256), 0, stream>>>(kh, qT, vh, tmp, T, C, scale);

    // 5) inverse rotation -> bf16, final projection -> f32 d_out
    rot_kernel<<<rgrid, dim3(256), 0, stream>>>(tmp, cosA, sinA, ah, npairs, 1);
    gemm_bias_kernel<<<ggrid, dim3(256), 0, stream>>>(ah, Wpt, bp, (float*)d_out, (int)M, C, C);
}
